// FastBEV_export_model_56375740727889
// MI455X (gfx1250) — compile-verified, matrix operand order checked by reading, not run on hardware
//
#include <hip/hip_runtime.h>
#include <stdint.h>

typedef __attribute__((ext_vector_type(16))) __bf16    v16bf;
typedef __attribute__((ext_vector_type(8)))  float     v8f;
typedef __attribute__((ext_vector_type(4)))  uint32_t  u32x4;

#define NIMG   6
#define C_IN   64
#define H_     64
#define W_     176
#define NXg    256
#define NYg    256
#define NZg    8
#define CH     512            // C_IN * NZ
#define CNECK  96
#define NOUT   10
#define PADW   258            // 256 + halo
#define VTOT   (NXg*NYg*NZg)  // 524288

// A staging: 128 channels (4 WMMA k-blocks) per stage, double buffered.
// Row = 256B data + 16B pad -> 272B stride (16B aligned; 68*r mod 64 distinct for r<16)
#define LDS_ROWB 272
#define LDS_BUFB (CNECK*LDS_ROWB)   // 26112 B per buffer, 2 buffers = 52224 B
#define NSTAGE   36                 // 9 taps x 4 stages (128 ch each)

#define BEVT_HALVES ((size_t)PADW*PADW*CH)          // 34,080,768 halves = 68,161,536 B
#define OFF_WA      (BEVT_HALVES*2)                 // bytes
#define OFF_WHA     (OFF_WA + (size_t)9*CNECK*CH*2) // bytes

union Frag  { v16bf v; u32x4 q[2]; };
union FragF { v8f v; float f[8]; };

#define WAIT_ASYNC(N) asm volatile("s_wait_asynccnt " #N ::: "memory")
#define WAIT_DS0()    asm volatile("s_wait_dscnt 0"      ::: "memory")

// ---------------- prep: zero padded bevT ----------------
__global__ void zero_bevt(u32x4* __restrict__ p, int n16) {
    int i = blockIdx.x * blockDim.x + threadIdx.x;
    u32x4 z = {0u, 0u, 0u, 0u};
    if (i < n16) p[i] = z;
}

// ---------------- prep: gather backprojection -> bevT (bf16) ----------------
__global__ void gather_bev(const float* __restrict__ feats,
                           const int*   __restrict__ coor,
                           __bf16*      __restrict__ bevT) {
    int v = blockIdx.x * blockDim.x + threadIdx.x;   // VTOT threads
    int z = v & 7;
    int y = (v >> 3) & 255;
    int x = v >> 11;
    int idx = coor[v];
    if (idx < 0 || idx >= NIMG * H_ * W_) return;    // sentinel row stays zero
    int n   = idx / (H_ * W_);
    int rem = idx - n * (H_ * W_);
    int h   = rem / W_;
    int w   = rem - h * W_;
    const float* src = feats + (((size_t)n * C_IN) * H_ + h) * W_ + w;
    __bf16* dst = bevT + ((size_t)((x + 1) * PADW + (y + 1))) * CH + z;
    #pragma unroll 8
    for (int c = 0; c < C_IN; ++c)
        dst[c * NZg] = (__bf16)src[(size_t)c * H_ * W_];  // ch = c*NZ + z
}

// ---------------- prep: repack weights to bf16 GEMM layout ----------------
__global__ void prep_weights(const float* __restrict__ wn,
                             const float* __restrict__ wh,
                             __bf16* __restrict__ wA,    // [tap][96][512]
                             __bf16* __restrict__ whA) { // [16][96], rows >=10 zero
    int i = blockIdx.x * blockDim.x + threadIdx.x;
    const int totalN = 9 * CNECK * CH;
    if (i < totalN) {
        int t  = i / (CNECK * CH);
        int r  = i - t * (CNECK * CH);
        int o  = r / CH;
        int ci = r - o * CH;
        wA[i] = (__bf16)wn[((size_t)o * CH + ci) * 9 + t];   // OIHW -> [t][o][ci]
    } else {
        int j = i - totalN;
        if (j < 16 * CNECK) {
            int o = j / CNECK;
            int k = j - o * CNECK;
            whA[j] = (o < NOUT) ? (__bf16)wh[o * CNECK + k] : (__bf16)0.0f;
        }
    }
}

// ---------------- main: implicit-GEMM neck conv + fused ReLU + 1x1 head ----------------
__global__ __launch_bounds__(256, 1)
void fastbev_neck_head(const __bf16* __restrict__ bevT,
                       const __bf16* __restrict__ wA,
                       const __bf16* __restrict__ whA,
                       const float*  __restrict__ bneck,
                       const float*  __restrict__ bhead,
                       float*        __restrict__ out) {
    __shared__ __align__(16) unsigned char ldsA[2 * LDS_BUFB];

    const int x    = blockIdx.x;          // output row
    const int tid  = threadIdx.x;
    const int wv   = tid >> 5;            // wave id 0..7 -> 32 output columns each
    const int lane = tid & 31;
    const int n    = lane & 15;           // N (and A-row M) index inside tile
    const int g    = lane >> 4;           // K half-group selector

    const uint32_t ldsbase = (uint32_t)(uintptr_t)&ldsA[0];

    // Async DMA of one 96x128ch A stage into LDS buffer `buf` (no VGPR transit).
    // 1536 b128 chunks over 256 threads = exactly 6 per thread (uniform, no branch).
    // chunk c = tid + i*256 -> row = (tid>>4)+i*16, seg = tid&15 (16B units).
    const int arow0 = tid >> 4;
    const int aseg  = tid & 15;
    auto issueA = [&](int buf, int blk) {               // blk = tap*4 + stage
        const __bf16* base = wA + (size_t)(blk >> 2) * (CNECK * CH) + (blk & 3) * 128;
        #pragma unroll
        for (int i = 0; i < 6; ++i) {
            int row = arow0 + i * 16;
            uint64_t ga = (uint64_t)(uintptr_t)(base + row * CH + aseg * 8);
            uint32_t la = ldsbase + buf * LDS_BUFB + row * LDS_ROWB + aseg * 16;
            asm volatile("global_load_async_to_lds_b128 %0, %1, off"
                         :: "v"(la), "v"(ga) : "memory");
        }
    };

    v8f acc[12];                          // [mt 0..5][nt 0..1] 16x16 f32 tiles
    const v8f vz8 = {0.f,0.f,0.f,0.f,0.f,0.f,0.f,0.f};
    #pragma unroll
    for (int i = 0; i < 12; ++i) acc[i] = vz8;

    const int ybase = wv * 32;

    issueA(0, 0);                                       // prologue: DMA stage 0
    int blk = 0;
    for (int kh = 0; kh < 3; ++kh) {
        const size_t prow = (size_t)(x + kh) * PADW;    // padded row (x+1)+(kh-1)
        for (int kw = 0; kw < 3; ++kw) {
            for (int kb2 = 0; kb2 < 4; ++kb2, ++blk) {  // 128-ch stage
                const int buf = blk & 1;
                if (blk + 1 < NSTAGE) {
                    issueA(buf ^ 1, blk + 1);           // DMA next stage (other buffer)
                    // in-order completion: cnt<=6 => current stage resident
                    WAIT_ASYNC(6);
                } else {
                    WAIT_ASYNC(0);
                }
                __syncthreads();                        // buf fully written by all waves

                // per-nt B row base for this (kh,kw,kb2)
                const __bf16* brow[2];
                #pragma unroll
                for (int nt = 0; nt < 2; ++nt) {
                    int yp = ybase + nt * 16 + n + kw;  // padded col (y+1)+(kw-1)
                    brow[nt] = bevT + (prow + yp) * CH + kb2 * 128 + g * 8;
                }
                const unsigned char* abuf = ldsA + buf * LDS_BUFB;

                #pragma unroll
                for (int kbsub = 0; kbsub < 4; ++kbsub) {   // 4 x 32-ch WMMA k-blocks
                    Frag bfr[2];
                    #pragma unroll
                    for (int nt = 0; nt < 2; ++nt) {
                        const u32x4* bp = (const u32x4*)(brow[nt] + kbsub * 32);
                        bfr[nt].q[0] = bp[0];               // K = g*8 .. g*8+7
                        bfr[nt].q[1] = bp[2];               // K = 16+g*8 .. +7
                    }
                    #pragma unroll
                    for (int mt = 0; mt < 6; ++mt) {
                        Frag af;
                        const unsigned char* arow =
                            abuf + (mt * 16 + n) * LDS_ROWB + kbsub * 64;
                        af.q[0] = *(const u32x4*)(arow + g * 16);
                        af.q[1] = *(const u32x4*)(arow + 32 + g * 16);
                        #pragma unroll
                        for (int nt = 0; nt < 2; ++nt) {
                            acc[mt * 2 + nt] = __builtin_amdgcn_wmma_f32_16x16x32_bf16(
                                false, af.v, false, bfr[nt].v,
                                (short)0, acc[mt * 2 + nt], false, false);
                        }
                    }
                }

                WAIT_DS0();                             // our LDS reads have landed
                __syncthreads();                        // safe to DMA-overwrite buf
            }
        }
    }

    // ---- fused epilogue: bias + ReLU + bf16, feed straight into head WMMA ----
    // C-layout lane (g,n) holds rows {8g..8g+7} of column n == exactly the
    // B-fragment K-runs required by the next GEMM. No shuffles needed.
    Frag ha[3];
    #pragma unroll
    for (int kbh = 0; kbh < 3; ++kbh) {
        const u32x4* hp = (const u32x4*)(whA + n * CNECK + kbh * 32 + g * 8);
        ha[kbh].q[0] = hp[0];
        ha[kbh].q[1] = hp[2];
    }
    float bn[6][8];
    #pragma unroll
    for (int mt = 0; mt < 6; ++mt)
        #pragma unroll
        for (int r = 0; r < 8; ++r)
            bn[mt][r] = bneck[mt * 16 + g * 8 + r];

    v8f hacc[2]; hacc[0] = vz8; hacc[1] = vz8;
    #pragma unroll
    for (int nt = 0; nt < 2; ++nt) {
        #pragma unroll
        for (int kbh = 0; kbh < 3; ++kbh) {
            Frag hb;
            FragF lo, hi;
            lo.v = acc[(2 * kbh) * 2 + nt];
            hi.v = acc[(2 * kbh + 1) * 2 + nt];
            #pragma unroll
            for (int e = 0; e < 8; ++e) {
                hb.v[e]     = (__bf16)fmaxf(lo.f[e] + bn[2 * kbh][e],     0.f);
                hb.v[e + 8] = (__bf16)fmaxf(hi.f[e] + bn[2 * kbh + 1][e], 0.f);
            }
            hacc[nt] = __builtin_amdgcn_wmma_f32_16x16x32_bf16(
                false, ha[kbh].v, false, hb.v, (short)0, hacc[nt], false, false);
        }
    }

    // write out[10][256][256] fp32
    #pragma unroll
    for (int nt = 0; nt < 2; ++nt) {
        int y = ybase + nt * 16 + n;
        FragF hf; hf.v = hacc[nt];
        #pragma unroll
        for (int r = 0; r < 8; ++r) {
            int o = g * 8 + r;
            if (o < NOUT)
                out[(size_t)o * (NXg * NYg) + x * NYg + y] = hf.f[r] + bhead[o];
        }
    }
}

extern "C" void kernel_launch(void* const* d_in, const int* in_sizes, int n_in,
                              void* d_out, int out_size, void* d_ws, size_t ws_size,
                              hipStream_t stream) {
    const float* feats = (const float*)d_in[0];
    const int*   coor  = (const int*)  d_in[1];
    const float* wn    = (const float*)d_in[2];
    const float* bn    = (const float*)d_in[3];
    const float* wh    = (const float*)d_in[4];
    const float* bh    = (const float*)d_in[5];
    float* out = (float*)d_out;

    char* ws = (char*)d_ws;
    __bf16* bevT = (__bf16*)ws;
    __bf16* wA   = (__bf16*)(ws + OFF_WA);
    __bf16* whA  = (__bf16*)(ws + OFF_WHA);

    // 1) zero padded bevT (border = SAME-padding zeros; interior overwritten below)
    int n16 = (int)(BEVT_HALVES / 8);                       // 4,260,096 uint4 stores
    zero_bevt<<<n16 / 256, 256, 0, stream>>>((u32x4*)ws, n16);

    // 2) gather backprojection into bf16 bevT
    gather_bev<<<VTOT / 256, 256, 0, stream>>>(feats, coor, bevT);

    // 3) weight repack to bf16 fragments
    int wtot = 9 * CNECK * CH + 16 * CNECK;                 // 443,904 (multiple of 256)
    prep_weights<<<wtot / 256, 256, 0, stream>>>(wn, wh, wA, whA);

    // 4) fused WMMA conv + ReLU + head (async double-buffered 128-ch A staging)
    fastbev_neck_head<<<NXg, 256, 0, stream>>>(bevT, wA, whA, bn, bh, out);
}
